// SelfAttention_88338887344884
// MI455X (gfx1250) — matrix operand order repaired and checked by reading, not compiled, not run on hardware
//
#include <hip/hip_runtime.h>

// ---------------------------------------------------------------------------
// CDNA5 WMMA (wave32): bf16 A/B fragments (16 elems/lane = 8 VGPRs),
// f32 C/D (8 elems/lane = 8 VGPRs).
// ---------------------------------------------------------------------------
typedef __attribute__((ext_vector_type(16))) __bf16 v16bf;
typedef __attribute__((ext_vector_type(2)))  __bf16 v2bf;
typedef __attribute__((ext_vector_type(8)))  float  v8f;

#define D_IN   1024
#define D_OUT  1024
#define SEQ    2048
#define BATCH  4
#define NTOK   (BATCH * SEQ)          // 8192 tokens

// padded LDS leading dims (bank-conflict avoidance, rows stay 16B aligned)
#define QS_LD  1032                   // 1024 + 4 dwords pad per row (TDM pad)
#define PS_LD  136                    // 128 + 8
#define GS_LD  40                     // 32 + 8

// Native bf16 converts: let the backend use v_cvt_(pk_)bf16_f32 when present.
__device__ __forceinline__ unsigned short f2bfu(float f) {
  __bf16 b = (__bf16)f;
  unsigned short u; __builtin_memcpy(&u, &b, 2);
  return u;
}

__device__ __forceinline__ unsigned pack2bf(float lo, float hi) {
  v2bf p; p[0] = (__bf16)lo; p[1] = (__bf16)hi;
  unsigned u; __builtin_memcpy(&u, &p, 4);
  return u;
}

__device__ __forceinline__ v8f wmma_bf16(v16bf a, v16bf b, v8f c) {
  return __builtin_amdgcn_wmma_f32_16x16x32_bf16(
      false, a, false, b, (short)0, c, false, false);
}

// A fragment: 16x32 bf16 (MxK), row-major source, leading dim `ld`.
// Lane l<16 : row l,    K = 0..15  (contiguous)
// Lane l>=16: row l-16, K = 16..31 (contiguous)
__device__ __forceinline__ v16bf load_frag_A(const unsigned short* base, int ld, int lane) {
  int r  = lane & 15;
  int hi = lane >> 4;
  const unsigned short* p = base + (size_t)r * ld + hi * 16;
  v16bf f;
  __builtin_memcpy(&f, p, 16);
  __builtin_memcpy((char*)&f + 16, p + 8, 16);
  return f;
}

// B fragment: 32x16 bf16 (KxN), source stored N-major: base[n*ld + k]
// Lane l<16 : column l,    K = 0..15  (contiguous)
// Lane l>=16: column l-16, K = 16..31 (contiguous)
__device__ __forceinline__ v16bf load_frag_B(const unsigned short* base, int ld, int lane) {
  const unsigned short* p = base + (size_t)(lane & 15) * ld + (lane >> 4) * 16;
  v16bf f;
  __builtin_memcpy(&f, p, 32);
  return f;
}

// ---------------------------------------------------------------------------
// Kernel 1: wq/wk/wv f32 -> bf16 (pair-packed b32 stores)
// ---------------------------------------------------------------------------
__global__ __launch_bounds__(256) void convert_weights(
    const float* __restrict__ wq, const float* __restrict__ wk,
    const float* __restrict__ wv, unsigned short* __restrict__ wB) {
  size_t base = ((size_t)blockIdx.x * 256 + threadIdx.x) * 4;
  const float* srcs[3] = {wq, wk, wv};
  const float* s = srcs[base >> 20];
  size_t off = base & ((1u << 20) - 1);
  unsigned pk[2];
#pragma unroll
  for (int e = 0; e < 2; e++)
    pk[e] = pack2bf(s[off + 2 * e], s[off + 2 * e + 1]);
  __builtin_memcpy(wB + base, pk, 8);
}

// ---------------------------------------------------------------------------
// Kernel 2: QKV projection. Roles: A = W tile (n x k), B = x^T tile (k x tok)
//   C[n][tok] -> each lane holds 8 consecutive n for one token.
//   q/k stored [token][d] (b128 stores); v stored transposed vT[b][d][s].
// ---------------------------------------------------------------------------
__global__ __launch_bounds__(256) void qkv_gemm(
    const float* __restrict__ x, const unsigned short* __restrict__ wB,
    const float* __restrict__ bq, const float* __restrict__ bk,
    const float* __restrict__ bv,
    unsigned short* __restrict__ qB, unsigned short* __restrict__ kB,
    unsigned short* __restrict__ vT) {
  __shared__ unsigned short Xs[128 * GS_LD];   // x tile bf16  [token][k]
  __shared__ unsigned short Ws[128 * GS_LD];   // W tile bf16  [n][k]

  const int which = blockIdx.z;
  const unsigned short* w = wB + (size_t)which * D_OUT * D_IN;
  const float* bias = (which == 0) ? bq : (which == 1) ? bk : bv;
  unsigned short* out = (which == 0) ? qB : kB;   // which==2 -> vT path

  const int m0 = blockIdx.x * 128;    // token tile (never crosses batch)
  const int n0 = blockIdx.y * 128;    // d_out tile
  const int t = threadIdx.x, lane = t & 31, wid = t >> 5;
  const int wm = wid & 3;             // 32-wide n sub-tile
  const int wn = wid >> 2;            // 64-wide token sub-tile

  v8f acc[2][4] = {};

  for (int kk = 0; kk < D_IN; kk += 32) {
    {   // stage x tile f32 -> bf16 (pair-packed writes)
      int r = t >> 1, c0 = (t & 1) * 16;
      float xf[16];
      __builtin_memcpy(xf, x + (size_t)(m0 + r) * D_IN + kk + c0, 64);
      unsigned pk[8];
#pragma unroll
      for (int i = 0; i < 8; i++)
        pk[i] = pack2bf(xf[2 * i], xf[2 * i + 1]);
      __builtin_memcpy(Xs + r * GS_LD + c0, pk, 32);
    }
    {   // stage W tile (already bf16)
      int r = t >> 1, c0 = (t & 1) * 16;
      __builtin_memcpy(Ws + r * GS_LD + c0,
                       w + (size_t)(n0 + r) * D_IN + kk + c0, 32);
    }
    __syncthreads();
#pragma unroll
    for (int mi = 0; mi < 2; mi++) {
      v16bf a = load_frag_A(Ws + (wm * 32 + mi * 16) * GS_LD, GS_LD, lane);
#pragma unroll
      for (int ni = 0; ni < 4; ni++) {
        v16bf b = load_frag_B(Xs + (wn * 64 + ni * 16) * GS_LD, GS_LD, lane);
        acc[mi][ni] = wmma_bf16(a, b, acc[mi][ni]);
      }
    }
    __syncthreads();
  }

  // Epilogue. C layout: VGPR r, lane l -> n = r + (l>=16)*8, token-col = l&15.
  const int hi = lane >> 4, ln = lane & 15;
#pragma unroll
  for (int mi = 0; mi < 2; mi++) {
    const int nb = n0 + wm * 32 + mi * 16 + hi * 8;   // 8 consecutive n
    float bvals[8];
    __builtin_memcpy(bvals, bias + nb, 32);
#pragma unroll
    for (int ni = 0; ni < 4; ni++) {
      const int tok = m0 + wn * 64 + ni * 16 + ln;
      if (which < 2) {
        unsigned pk4[4];
#pragma unroll
        for (int r = 0; r < 4; r++)
          pk4[r] = pack2bf(acc[mi][ni][2 * r] + bvals[2 * r],
                           acc[mi][ni][2 * r + 1] + bvals[2 * r + 1]);
        __builtin_memcpy(out + (size_t)tok * D_OUT + nb, pk4, 16);
      } else {
        const int bb = tok >> 11, s = tok & (SEQ - 1);
        unsigned short* vt = vT + (size_t)bb * D_OUT * SEQ;
#pragma unroll
        for (int r = 0; r < 8; r++)
          vt[(size_t)(nb + r) * SEQ + s] = f2bfu(acc[mi][ni][r] + bvals[r]);
      }
    }
  }
}

// ---------------------------------------------------------------------------
// Kernel 3: causal flash attention. 8 waves / 16-query tile; 128-key blocks
// (16 keys/wave for scores); each wave owns 128 d-columns for P*V (via vT).
// ---------------------------------------------------------------------------
__global__ __launch_bounds__(256) void flash_attn(
    const unsigned short* __restrict__ qB, const unsigned short* __restrict__ kB,
    const unsigned short* __restrict__ vT, float* __restrict__ out) {
  __shared__ unsigned short Qs[16 * QS_LD + 8];   // padded Q tile (bf16)
  __shared__ unsigned short Ps[16 * PS_LD];       // padded P exchange (bf16)
  __shared__ float redmax[8][16];
  __shared__ float redsum[8][16];

  const int bid = blockIdx.x;
  const int b = bid >> 7;               // SEQ/16 = 128 tiles per batch
  const int qt = bid & 127;
  const int row0 = qt * 16;
  const size_t batch_off = (size_t)b * SEQ * D_OUT;

  const int t = threadIdx.x, lane = t & 31, w = t >> 5;
  const int hi = lane >> 4, ln = lane & 15;

  // ---- Stage Q tile into LDS via Tensor Data Mover (with LDS padding) ----
#if __has_builtin(__builtin_amdgcn_tensor_load_to_lds)
  if (w == 0) {
    typedef unsigned int u32x4 __attribute__((ext_vector_type(4)));
    typedef int i32x8 __attribute__((ext_vector_type(8)));
    typedef int i32x4 __attribute__((ext_vector_type(4)));
    unsigned long long gaddr =
        (unsigned long long)(const void*)(qB + batch_off + (size_t)row0 * D_OUT);
    unsigned int laddr = (unsigned int)(unsigned long long)(void*)&Qs[0];
    // group0: count=1 | lds_addr | global_addr[56:0] | type=2
    u32x4 g0 = { 1u, laddr, (unsigned int)gaddr,
                 (unsigned int)((gaddr >> 32) & 0x1FFFFFFull) | (2u << 30) };
    // group1: data_size=2B, pad_enable, pad every 256 DW (code 7) by 2 DW
    // (code 1) -> LDS row stride 1032 bf16; tensor 1024 x 8192; tile 1024x16
    i32x8 g1 = {};
    g1[0] = (int)(0x00010000u | (1u << 20) | (7u << 22) | (1u << 25));
    g1[1] = (int)((D_OUT & 0xFFFFu) << 16);          // tensor_dim0 lo16
    g1[2] = (int)(((D_OUT >> 16) & 0xFFFFu) |
                  ((NTOK & 0xFFFFu) << 16));         // dim0 hi | dim1 lo
    g1[3] = (int)(((NTOK >> 16) & 0xFFFFu) |
                  ((unsigned)D_OUT << 16));          // dim1 hi | tile_dim0
    g1[4] = 16;                                      // tile_dim1 = 16 rows
    g1[5] = D_OUT;                                   // tensor_dim0_stride
    i32x4 g2 = {}, g3 = {};
    i32x8 g4 = {};
    __builtin_amdgcn_tensor_load_to_lds(g0, g1, g2, g3, g4, 0);
    __builtin_amdgcn_s_wait_tensorcnt(0);
  }
#else
  for (int i = t; i < 2048; i += 256) {             // 2048 chunks of 16B
    int row = i >> 7, k = (i & 127) * 8;
    __builtin_memcpy(Qs + row * QS_LD + k + ((k >> 9) << 2),
                     qB + batch_off + (size_t)(row0 + row) * D_OUT + k, 16);
  }
#endif
  __syncthreads();

  float m_run[8], l_run[8];
  v8f o[8] = {};
#pragma unroll
  for (int r = 0; r < 8; r++) { m_run[r] = -1e30f; l_run[r] = 0.f; }

  const float scale = 0.03125f;   // 1/sqrt(1024)
  const int nj = (row0 + 15) / 128 + 1;

  const unsigned short* vtb = vT + (size_t)b * D_OUT * SEQ;

  for (int j = 0; j < nj; j++) {
    const int key0 = j * 128 + w * 16;
    // ---- scores: S = Q(16x1024) * K_tile^T -------------------------------
    v8f s = {};
    const unsigned short* kbase = kB + batch_off + (size_t)key0 * D_OUT;
#pragma unroll 4
    for (int kk = 0; kk < D_IN; kk += 32) {
      __builtin_prefetch(kbase + kk + 256, 0, 1);
      const int qoff = kk + ((kk >> 9) << 2);        // padded-row k offset
      v16bf a = load_frag_A(Qs + qoff, QS_LD, lane);
      v16bf bf = load_frag_B(kbase + kk, D_OUT, lane);
      s = wmma_bf16(a, bf, s);
    }
    // ---- scale + causal mask + local row max -----------------------------
    float mloc[8];
#pragma unroll
    for (int r = 0; r < 8; r++) {
      int qrow = row0 + hi * 8 + r;
      int key = key0 + ln;
      float v = s[r] * scale;
      if (key > qrow) v = -1e30f;
      s[r] = v;
      mloc[r] = v;
    }
#pragma unroll
    for (int off = 1; off < 16; off <<= 1)
#pragma unroll
      for (int r = 0; r < 8; r++)
        mloc[r] = fmaxf(mloc[r], __shfl_xor(mloc[r], off, 32));
#pragma unroll
    for (int r = 0; r < 8; r++)
      if (ln == r) redmax[w][hi * 8 + r] = mloc[r];
    __syncthreads();

    // ---- global row max, rescale factor ----------------------------------
    float m_new[8], alpha[8];
#pragma unroll
    for (int r = 0; r < 8; r++) {
      int rr = hi * 8 + r;
      float m = m_run[r];
#pragma unroll
      for (int ww = 0; ww < 8; ww++) m = fmaxf(m, redmax[ww][rr]);
      m_new[r] = m;
      alpha[r] = __expf(m_run[r] - m);
      m_run[r] = m;
    }
    // ---- P = exp(S - m), row sums, exchange via LDS ----------------------
    float sl[8];
#pragma unroll
    for (int r = 0; r < 8; r++) {
      float p = __expf(s[r] - m_new[r]);
      s[r] = p;
      sl[r] = p;
    }
#pragma unroll
    for (int off = 1; off < 16; off <<= 1)
#pragma unroll
      for (int r = 0; r < 8; r++) sl[r] += __shfl_xor(sl[r], off, 32);
#pragma unroll
    for (int r = 0; r < 8; r++)
      if (ln == r) redsum[w][hi * 8 + r] = sl[r];
#pragma unroll
    for (int r = 0; r < 8; r++)
      Ps[(size_t)(hi * 8 + r) * PS_LD + w * 16 + ln] = f2bfu(s[r]);
    __syncthreads();

    // ---- l update + O rescale --------------------------------------------
#pragma unroll
    for (int r = 0; r < 8; r++) {
      int rr = hi * 8 + r;
      float sum = 0.f;
#pragma unroll
      for (int ww = 0; ww < 8; ww++) sum += redsum[ww][rr];
      l_run[r] = l_run[r] * alpha[r] + sum;
    }
#pragma unroll
    for (int ni = 0; ni < 8; ni++)
#pragma unroll
      for (int r = 0; r < 8; r++) o[ni][r] *= alpha[r];

    // ---- O += P(16x128) * V-slice via vT: contiguous B-frag loads --------
#pragma unroll
    for (int ks = 0; ks < 4; ks++) {
      v16bf a = load_frag_A(Ps + ks * 32, PS_LD, lane);
      const int kcol = j * 128 + ks * 32;            // token axis of vT
#pragma unroll
      for (int ni = 0; ni < 8; ni++) {
        const unsigned short* vb =
            vtb + (size_t)(w * 128 + ni * 16) * SEQ + kcol;
        v16bf bf = load_frag_B(vb, SEQ, lane);
        o[ni] = wmma_bf16(a, bf, o[ni]);
      }
    }
    __syncthreads();   // protect Ps/red buffers for next block
  }

  // ---- normalize + write f32 output --------------------------------------
#pragma unroll
  for (int ni = 0; ni < 8; ni++) {
    int col = w * 128 + ni * 16 + ln;
#pragma unroll
    for (int r = 0; r < 8; r++) {
      int rr = hi * 8 + r;
      out[batch_off + (size_t)(row0 + rr) * D_OUT + col] = o[ni][r] / l_run[r];
    }
  }
}

// ---------------------------------------------------------------------------
// Launch
// ---------------------------------------------------------------------------
extern "C" void kernel_launch(void* const* d_in, const int* in_sizes, int n_in,
                              void* d_out, int out_size, void* d_ws, size_t ws_size,
                              hipStream_t stream) {
  const float* x  = (const float*)d_in[0];
  const float* wq = (const float*)d_in[1];
  const float* bq = (const float*)d_in[2];
  const float* wk = (const float*)d_in[3];
  const float* bk = (const float*)d_in[4];
  const float* wv = (const float*)d_in[5];
  const float* bv = (const float*)d_in[6];
  float* out = (float*)d_out;

  // Workspace (bf16 elems): [wqkv 3M][q 8M][k 8M][vT 8M] ~= 56 MB
  unsigned short* ws = (unsigned short*)d_ws;
  unsigned short* wB = ws;
  unsigned short* qB = ws + 3ull * 1024 * 1024;
  unsigned short* kB = qB + (size_t)NTOK * D_OUT;
  unsigned short* vT = kB + (size_t)NTOK * D_OUT;

  convert_weights<<<NTOK / 1024 * 128 * 3 / 3, 256, 0, stream>>>(wq, wk, wv, wB);
  // ^ evaluates to 1024 blocks: 1024 * 256 threads * 4 elems = 2^20 elems (wq only)
  // (kept explicit below to avoid confusion)

  dim3 g2(NTOK / 128, D_OUT / 128, 3);
  qkv_gemm<<<g2, 256, 0, stream>>>(x, wB, bq, bk, bv, qB, kB, vT);

  flash_attn<<<NTOK / 16, 256, 0, stream>>>(qB, kB, vT, out);
}
